// DecomposeBlock_13984413516299
// MI455X (gfx1250) — matrix operand order repaired and checked by reading, not compiled, not run on hardware
//
#include <hip/hip_runtime.h>
#include <hip/hip_bf16.h>

#define FUTURE_LEN 12

typedef __attribute__((ext_vector_type(16))) _Float16 v16h;
typedef __attribute__((ext_vector_type(8)))  _Float16 v8h;
typedef __attribute__((ext_vector_type(8)))  float    v8f;

// ---------------- input indices (setup_inputs insertion order, params dict
// flattened in insertion order) ----------------
enum {
  IN_F = 0, IN_LOCS, IN_SHOT, IN_HIT, IN_NA, IN_NS, IN_FC, IN_TF,
  IN_GATW, IN_GATAL, IN_GATAR,
  IN_T0W1, IN_T0B1, IN_T0W2, IN_T0B2, IN_T0WD, IN_T0BD,
  IN_T1W1, IN_T1B1, IN_T1W2, IN_T1B2,
  IN_T2W1, IN_T2B1, IN_T2W2, IN_T2B2, IN_T2WD, IN_T2BD,
  IN_DYW0, IN_DYB0, IN_DYW1, IN_DYB1, IN_DYW2, IN_DYB2,
  IN_DHW0, IN_DHB0, IN_DHW1, IN_DHB1, IN_DHW2, IN_DHB2,
  IN_DSW0, IN_DSB0, IN_DSW1, IN_DSB1, IN_DSW2, IN_DSB2
};

__device__ __forceinline__ float wave_sum32(float v) {
#pragma unroll
  for (int off = 16; off > 0; off >>= 1) v += __shfl_xor(v, off, 32);
  return v;
}

// ---------------- weight packing into WMMA B-fragment layout ----------------
// Packed layout: [ntile][ktile][lane(32)][e(16)] halves, contiguous.
// Per ISA: column n = lane&15, k = (lane>>4)*8 + e for e<8, else 16 + (lane>>4)*8 + (e-8)  (16-bit B, 32x16 tile).
// mode 0: packed row k -> src row k if k < Ksrc else 0
// mode 1 (decoder_hit W0): X cols are [feat 0..83 | shot 84..99 | hit 100..103];
//         dh weight rows are [feat 0..83 | hit 84..87].
__global__ void __launch_bounds__(256)
pack_wmma_b(const float* __restrict__ W, _Float16* __restrict__ Wp,
            int Kpad, int Npad, int Ksrc, int Nsrc, int mode)
{
  int idx = blockIdx.x * blockDim.x + threadIdx.x;
  int total = Kpad * Npad;
  if (idx >= total) return;
  int e    = idx & 15;
  int lane = (idx >> 4) & 31;
  int rest = idx >> 9;          // = nt*nKT + kt
  int nKT  = Kpad >> 5;
  int kt   = rest % nKT;
  int nt   = rest / nKT;
  int k = kt * 32 + ((e < 8) ? (lane >> 4) * 8 + e : 16 + (lane >> 4) * 8 + (e - 8));
  int n = nt * 16 + (lane & 15);
  int srcRow;
  if (mode == 0) {
    srcRow = (k < Ksrc) ? k : -1;
  } else {
    if (k < 84)                   srcRow = k;
    else if (k >= 100 && k < 104) srcRow = 84 + (k - 100);
    else                          srcRow = -1;
  }
  float v = (srcRow >= 0 && n < Nsrc) ? W[(size_t)srcRow * Nsrc + n] : 0.0f;
  Wp[idx] = (_Float16)v;
}

// ---------------- recurrent state init (jnp.repeat over samples) ----------------
__global__ void __launch_bounds__(256)
state_init(const float* __restrict__ cl, const float* __restrict__ cs,
           const float* __restrict__ ch, float* __restrict__ locs,
           float* __restrict__ shot, float* __restrict__ hit, int N, int S)
{
  int r = blockIdx.x * blockDim.x + threadIdx.x;
  if (r >= N) return;
  int b = r / S;
#pragma unroll
  for (int c = 0; c < 4;  ++c) locs[(size_t)r * 4  + c] = cl[(size_t)b * 4  + c];
#pragma unroll
  for (int c = 0; c < 16; ++c) shot[(size_t)r * 16 + c] = cs[(size_t)b * 16 + c];
#pragma unroll
  for (int c = 0; c < 4;  ++c) hit [(size_t)r * 4  + c] = ch[(size_t)b * 4  + c];
}

// ---------------- fused GAT + TCN + X assembly (one wave32 per group) ----------
struct PrepArgs {
  const float *f, *locs, *shot, *hit;
  const float *gatW, *gatAl, *gatAr;
  const float *t0w1, *t0b1, *t0w2, *t0b2, *t0wd, *t0bd;
  const float *t1w1, *t1b1, *t1w2, *t1b2;
  const float *t2w1, *t2b1, *t2w2, *t2b2, *t2wd, *t2bd;
  _Float16* X;
  int G;
};

__global__ void __launch_bounds__(256) prepare_kernel(PrepArgs A)
{
  __shared__ float s_inp[8][4][20];   // inp rows per agent
  __shared__ float s_A[8][32][4];     // channel x time scratch
  __shared__ float s_B[8][32][4];
  const int w    = threadIdx.x >> 5;
  const int lane = threadIdx.x & 31;
  const int g    = blockIdx.x * 8 + w;
  const bool act = (g < A.G);

  if (act) {
    // inp = concat(f, locs) -> LDS + X cols 0..19 (f16)
    for (int idx = lane; idx < 80; idx += 32) {
      int i = idx / 20, k = idx % 20;
      int r = g * 4 + i;
      float v = (k < 16) ? A.f[(size_t)r * 16 + k] : A.locs[(size_t)r * 4 + (k - 16)];
      s_inp[w][i][k] = v;
      A.X[(size_t)r * 128 + k] = (_Float16)v;
    }
    // X cols 84..99 = shot, 100..103 = hit, 104..127 = 0
    for (int idx = lane; idx < 176; idx += 32) {
      int i = idx / 44, c = idx % 44;
      int r = g * 4 + i;
      float v;
      if (c < 16)      v = A.shot[(size_t)r * 16 + c];
      else if (c < 20) v = A.hit[(size_t)r * 4 + (c - 16)];
      else             v = 0.0f;
      A.X[(size_t)r * 128 + 84 + c] = (_Float16)v;
    }
  }
  __syncthreads();

  const int c = lane;   // feature / channel index

  // ---------------- GAT: lane owns output feature c of z (4 x 32) ----------
  if (act) {
    float z[4];
#pragma unroll
    for (int i = 0; i < 4; ++i) {
      float acc = 0.0f;
      for (int k = 0; k < 20; ++k) acc += s_inp[w][i][k] * A.gatW[k * 32 + c];
      z[i] = acc;
    }
    float alc = A.gatAl[c], arc = A.gatAr[c];
    float el[4], er[4];
#pragma unroll
    for (int i = 0; i < 4; ++i) {
      el[i] = wave_sum32(z[i] * alc);
      er[i] = wave_sum32(z[i] * arc);
    }
    float e[4][4];
#pragma unroll
    for (int i = 0; i < 4; ++i)
#pragma unroll
      for (int j = 0; j < 4; ++j) {
        float v = el[i] + er[j];
        v = (v > 0.0f) ? v : 0.2f * v;        // leaky relu
        e[i][j] = (i == j) ? -1e30f : v;      // mask self loop
      }
#pragma unroll
    for (int j = 0; j < 4; ++j) {             // softmax over sources i per dst j
      float m = -1e30f;
#pragma unroll
      for (int i = 0; i < 4; ++i) m = fmaxf(m, e[i][j]);
      float s = 0.0f, ex[4];
#pragma unroll
      for (int i = 0; i < 4; ++i) { ex[i] = __expf(e[i][j] - m); s += ex[i]; }
      float inv = 1.0f / s, o = 0.0f;
#pragma unroll
      for (int i = 0; i < 4; ++i) o += ex[i] * inv * z[i];
      A.X[(size_t)(g * 4 + j) * 128 + 20 + c] = (_Float16)o;
    }
  }

  // ---------------- TCN (channels x L=4, causal dilated convs) -------------
  // block0: 20 -> 16, dil 1, with 1x1 downsample
  if (act && c < 16) {
    for (int l = 0; l < 4; ++l) {
      float acc = A.t0b1[c];
      for (int ci = 0; ci < 20; ++ci) {
        float xm = (l >= 1) ? s_inp[w][l - 1][ci] : 0.0f;
        acc += A.t0w1[(c * 20 + ci) * 2 + 0] * xm
             + A.t0w1[(c * 20 + ci) * 2 + 1] * s_inp[w][l][ci];
      }
      s_A[w][c][l] = fmaxf(acc, 0.0f);
    }
  }
  __syncthreads();
  if (act && c < 16) {
    for (int l = 0; l < 4; ++l) {
      float acc = A.t0b2[c];
      for (int ci = 0; ci < 16; ++ci) {
        float xm = (l >= 1) ? s_A[w][ci][l - 1] : 0.0f;
        acc += A.t0w2[(c * 16 + ci) * 2 + 0] * xm
             + A.t0w2[(c * 16 + ci) * 2 + 1] * s_A[w][ci][l];
      }
      float h2 = fmaxf(acc, 0.0f);
      float res = A.t0bd[c];
      for (int ci = 0; ci < 20; ++ci) res += A.t0wd[c * 20 + ci] * s_inp[w][l][ci];
      s_B[w][c][l] = fmaxf(h2 + res, 0.0f);
    }
  }
  __syncthreads();
  // block1: 16 -> 16, dil 2, identity residual
  if (act && c < 16) {
    for (int l = 0; l < 4; ++l) {
      float acc = A.t1b1[c];
      for (int ci = 0; ci < 16; ++ci) {
        float xm = (l >= 2) ? s_B[w][ci][l - 2] : 0.0f;
        acc += A.t1w1[(c * 16 + ci) * 2 + 0] * xm
             + A.t1w1[(c * 16 + ci) * 2 + 1] * s_B[w][ci][l];
      }
      s_A[w][c][l] = fmaxf(acc, 0.0f);
    }
  }
  __syncthreads();
  if (act && c < 16) {
    float o[4];
    for (int l = 0; l < 4; ++l) {
      float acc = A.t1b2[c];
      for (int ci = 0; ci < 16; ++ci) {
        float xm = (l >= 2) ? s_A[w][ci][l - 2] : 0.0f;
        acc += A.t1w2[(c * 16 + ci) * 2 + 0] * xm
             + A.t1w2[(c * 16 + ci) * 2 + 1] * s_A[w][ci][l];
      }
      o[l] = fmaxf(fmaxf(acc, 0.0f) + s_B[w][c][l], 0.0f);
    }
    for (int l = 0; l < 4; ++l) s_B[w][c][l] = o[l];
  }
  __syncthreads();
  // block2: 16 -> 32, dil 4 (taps at l-4 always padding for L=4), 1x1 down
  if (act) {
    for (int l = 0; l < 4; ++l) {
      float acc = A.t2b1[c];
      for (int ci = 0; ci < 16; ++ci)
        acc += A.t2w1[(c * 16 + ci) * 2 + 1] * s_B[w][ci][l];
      s_A[w][c][l] = fmaxf(acc, 0.0f);
    }
  }
  __syncthreads();
  if (act) {
    for (int l = 0; l < 4; ++l) {
      float acc = A.t2b2[c];
      for (int ci = 0; ci < 32; ++ci)
        acc += A.t2w2[(c * 32 + ci) * 2 + 1] * s_A[w][ci][l];
      float res = A.t2bd[c];
      for (int ci = 0; ci < 16; ++ci) res += A.t2wd[c * 16 + ci] * s_B[w][ci][l];
      float o = fmaxf(fmaxf(acc, 0.0f) + res, 0.0f);
      // raw (G,32,4) -> (N,32) reshape: row j=c/8 gets (channel 8j+q, time l) at q*4+l
      A.X[(size_t)(g * 4 + (c >> 3)) * 128 + 52 + (c & 7) * 4 + l] = (_Float16)o;
    }
  }
}

// ---------------- WMMA GEMM: Y = act(X @ Wp + bias) --------------------------
// X:   M x K row-major f16; Wp: packed B fragments [ntile][ktile][lane][16].
// K / Nout are compile-time so every fragment load is base + immediate offset
// and the kt loop fully unrolls (A fragments held in registers, reused over NT
// B tiles). One wave computes a 16 x (16*NT) output tile; f32 accumulate.
template <int K, int Nout, int NT, bool RELU, bool OUT_F16>
__global__ void __launch_bounds__(256)
gemm_wmma(const _Float16* __restrict__ X, const _Float16* __restrict__ Wp,
          const float* __restrict__ bias, void* __restrict__ Yv,
          int M, int Nreal)
{
  constexpr int nKT     = K / 32;
  constexpr int nGroups = Nout / (16 * NT);

  const int lane = threadIdx.x & 31;
  const int hlf  = lane >> 4;
  const int wib  = threadIdx.x >> 5;
  const int wpb  = blockDim.x >> 5;
  const int mTiles = M >> 4;
  const int total  = mTiles * nGroups;

  for (int task = blockIdx.x * wpb + wib; task < total; task += gridDim.x * wpb) {
    const int mt = task / nGroups;
    const int ng = task % nGroups;

    // A per-lane layout: row m = lane&15; elements 0..7 -> K=hlf*8+0..7,
    // elements 8..15 -> K=16+hlf*8+0..7 (two contiguous 16B chunks).
    const _Float16* arow = X + (size_t)(mt * 16 + (lane & 15)) * K + hlf * 8;
    // B fragments: per-lane 32B chunk; stride 512 halves per k-tile.
    const _Float16* brow = Wp + ((size_t)(ng * NT) * nKT * 32 + (size_t)lane) * 16;

    v16h afrag[nKT];
#pragma unroll
    for (int kt = 0; kt < nKT; ++kt) {
      v8h a0 = *(const v8h*)(arow + kt * 32);
      v8h a1 = *(const v8h*)(arow + kt * 32 + 16);
      afrag[kt] = __builtin_shufflevector(a0, a1, 0, 1, 2, 3, 4, 5, 6, 7,
                                          8, 9, 10, 11, 12, 13, 14, 15);
    }

    v8f acc[NT];
#pragma unroll
    for (int t = 0; t < NT; ++t) {
#pragma unroll
      for (int j = 0; j < 8; ++j) acc[t][j] = 0.0f;
    }

#pragma unroll
    for (int t = 0; t < NT; ++t) {
#pragma unroll
      for (int kt = 0; kt < nKT; ++kt) {
        const _Float16* bp = brow + (size_t)(t * nKT + kt) * 512;
        v8h b0 = *(const v8h*)(bp);
        v8h b1 = *(const v8h*)(bp + 8);
        v16h b = __builtin_shufflevector(b0, b1, 0, 1, 2, 3, 4, 5, 6, 7,
                                         8, 9, 10, 11, 12, 13, 14, 15);
        acc[t] = __builtin_amdgcn_wmma_f32_16x16x32_f16(
            false, afrag[kt], false, b, (short)0, acc[t], false, false);
      }
    }

    // D layout: n = lane&15, m = j + 8*hlf
#pragma unroll
    for (int t = 0; t < NT; ++t) {
      int n = (ng * NT + t) * 16 + (lane & 15);
      float bn = (n < Nreal) ? bias[n] : 0.0f;
#pragma unroll
      for (int j = 0; j < 8; ++j) {
        int m = mt * 16 + hlf * 8 + j;
        float v = acc[t][j] + bn;
        if (RELU) v = fmaxf(v, 0.0f);
        if (OUT_F16) {
          ((_Float16*)Yv)[(size_t)m * Nout + n] = (_Float16)v;
        } else if (n < Nreal) {
          ((float*)Yv)[(size_t)m * Nout + n] = v;
        }
      }
    }
  }
}

// ---------------- epilogues ----------------
__global__ void __launch_bounds__(256)
epi_y(const float* __restrict__ H3, float* __restrict__ locs,
      float* __restrict__ outY, int N, int t)
{
  int r = blockIdx.x * blockDim.x + threadIdx.x;
  if (r >= N) return;
#pragma unroll
  for (int c = 0; c < 4; ++c) {
    float v = H3[(size_t)r * 16 + c] + locs[(size_t)r * 4 + c];
    locs[(size_t)r * 4 + c] = v;
    if (c < 2) outY[((size_t)r * FUTURE_LEN + t) * 2 + c] = v;
  }
}

__global__ void __launch_bounds__(256)
epi_soft(const float* __restrict__ H3, float* __restrict__ state,
         float* __restrict__ outB, int N, int t, int C)
{
  int r = blockIdx.x * blockDim.x + threadIdx.x;
  if (r >= N) return;
  float m = -1e30f;
  for (int c = 0; c < C; ++c) m = fmaxf(m, H3[(size_t)r * 16 + c]);
  float e[16], s = 0.0f;
  for (int c = 0; c < C; ++c) { e[c] = __expf(H3[(size_t)r * 16 + c] - m); s += e[c]; }
  float inv = 1.0f / s;
  for (int c = 0; c < C; ++c) {
    float p = e[c] * inv;
    state[(size_t)r * C + c] = p;
    outB[((size_t)r * FUTURE_LEN + t) * C + c] = p;
  }
}

// ---------------- host ----------------
extern "C" void kernel_launch(void* const* d_in, const int* in_sizes, int n_in,
                              void* d_out, int out_size, void* d_ws, size_t ws_size,
                              hipStream_t stream)
{
  const int N  = in_sizes[IN_F] / 16;      // rows (B0*S)
  const int B0 = in_sizes[IN_LOCS] / 4;
  const int S  = N / B0;                   // num_samples
  const int G  = N / 4;                    // groups of 4 agents

  // ---- workspace carve (256B aligned) ----
  char* wsp = (char*)d_ws;
  auto take = [&](size_t bytes) -> void* {
    void* p = (void*)wsp;
    wsp += (bytes + 255) & ~(size_t)255;
    return p;
  };
  _Float16* X    = (_Float16*)take((size_t)N * 128 * sizeof(_Float16));
  _Float16* H1   = (_Float16*)take((size_t)N * 256 * sizeof(_Float16));
  _Float16* H2   = (_Float16*)take((size_t)N * 128 * sizeof(_Float16));
  float*    H3   = (float*)   take((size_t)N * 16  * sizeof(float));
  float*    locs = (float*)   take((size_t)N * 4   * sizeof(float));
  float*    shot = (float*)   take((size_t)N * 16  * sizeof(float));
  float*    hit  = (float*)   take((size_t)N * 4   * sizeof(float));
  _Float16* W0p[3]; _Float16* W1p[3]; _Float16* W2p[3];
  for (int d = 0; d < 3; ++d) {
    W0p[d] = (_Float16*)take((size_t)128 * 256 * sizeof(_Float16));
    W1p[d] = (_Float16*)take((size_t)256 * 128 * sizeof(_Float16));
    W2p[d] = (_Float16*)take((size_t)128 * 16  * sizeof(_Float16));
  }

  const float* W0s[3] = {(const float*)d_in[IN_DYW0], (const float*)d_in[IN_DHW0], (const float*)d_in[IN_DSW0]};
  const float* B0s[3] = {(const float*)d_in[IN_DYB0], (const float*)d_in[IN_DHB0], (const float*)d_in[IN_DSB0]};
  const float* W1s[3] = {(const float*)d_in[IN_DYW1], (const float*)d_in[IN_DHW1], (const float*)d_in[IN_DSW1]};
  const float* B1s[3] = {(const float*)d_in[IN_DYB1], (const float*)d_in[IN_DHB1], (const float*)d_in[IN_DSB1]};
  const float* W2s[3] = {(const float*)d_in[IN_DYW2], (const float*)d_in[IN_DHW2], (const float*)d_in[IN_DSW2]};
  const float* B2s[3] = {(const float*)d_in[IN_DYB2], (const float*)d_in[IN_DHB2], (const float*)d_in[IN_DSB2]};
  const int K0src[3]  = {84, 88, 104};
  const int mode0[3]  = {0, 1, 0};
  const int N2real[3] = {4, 4, 16};

  // ---- pack decoder weights to f16 WMMA B fragments (per launch, deterministic) ----
  for (int d = 0; d < 3; ++d) {
    pack_wmma_b<<<(128 * 256 + 255) / 256, 256, 0, stream>>>(W0s[d], W0p[d], 128, 256, K0src[d], 256, mode0[d]);
    pack_wmma_b<<<(256 * 128 + 255) / 256, 256, 0, stream>>>(W1s[d], W1p[d], 256, 128, 256, 128, 0);
    pack_wmma_b<<<(128 * 16  + 255) / 256, 256, 0, stream>>>(W2s[d], W2p[d], 128, 16, 128, N2real[d], 0);
  }

  // ---- init recurrent state ----
  state_init<<<(N + 255) / 256, 256, 0, stream>>>(
      (const float*)d_in[IN_LOCS], (const float*)d_in[IN_SHOT],
      (const float*)d_in[IN_HIT], locs, shot, hit, N, S);

  float* out  = (float*)d_out;
  float* outY = out;                                  // (N,12,2)
  float* outS = out + (size_t)N * FUTURE_LEN * 2;     // (N,12,16)
  float* outP = outS + (size_t)N * FUTURE_LEN * 16;   // (N,12,4)

  PrepArgs PA;
  PA.f = (const float*)d_in[IN_F];
  PA.locs = locs; PA.shot = shot; PA.hit = hit;
  PA.gatW = (const float*)d_in[IN_GATW];
  PA.gatAl = (const float*)d_in[IN_GATAL];
  PA.gatAr = (const float*)d_in[IN_GATAR];
  PA.t0w1 = (const float*)d_in[IN_T0W1]; PA.t0b1 = (const float*)d_in[IN_T0B1];
  PA.t0w2 = (const float*)d_in[IN_T0W2]; PA.t0b2 = (const float*)d_in[IN_T0B2];
  PA.t0wd = (const float*)d_in[IN_T0WD]; PA.t0bd = (const float*)d_in[IN_T0BD];
  PA.t1w1 = (const float*)d_in[IN_T1W1]; PA.t1b1 = (const float*)d_in[IN_T1B1];
  PA.t1w2 = (const float*)d_in[IN_T1W2]; PA.t1b2 = (const float*)d_in[IN_T1B2];
  PA.t2w1 = (const float*)d_in[IN_T2W1]; PA.t2b1 = (const float*)d_in[IN_T2B1];
  PA.t2w2 = (const float*)d_in[IN_T2W2]; PA.t2b2 = (const float*)d_in[IN_T2B2];
  PA.t2wd = (const float*)d_in[IN_T2WD]; PA.t2bd = (const float*)d_in[IN_T2BD];
  PA.X = X; PA.G = G;

  const int tasks1 = (N / 16) * (256 / 64);  // layer1 wave-tiles (NT=4)
  const int tasks2 = (N / 16) * (128 / 64);  // layer2 wave-tiles (NT=4)
  const int tasks3 = (N / 16);               // layer3 wave-tiles (NT=1)

  for (int t = 0; t < FUTURE_LEN; ++t) {
    prepare_kernel<<<(G + 7) / 8, 256, 0, stream>>>(PA);
    for (int d = 0; d < 3; ++d) {
      gemm_wmma<128, 256, 4, true, true><<<(tasks1 + 7) / 8, 256, 0, stream>>>(
          X, W0p[d], B0s[d], (void*)H1, N, 256);
      gemm_wmma<256, 128, 4, true, true><<<(tasks2 + 7) / 8, 256, 0, stream>>>(
          H1, W1p[d], B1s[d], (void*)H2, N, 128);
      gemm_wmma<128, 16, 1, false, false><<<(tasks3 + 7) / 8, 256, 0, stream>>>(
          H2, W2p[d], B2s[d], (void*)H3, N, N2real[d]);
      if (d == 0)
        epi_y<<<(N + 255) / 256, 256, 0, stream>>>(H3, locs, outY, N, t);
      else if (d == 1)
        epi_soft<<<(N + 255) / 256, 256, 0, stream>>>(H3, hit, outP, N, t, 4);
      else
        epi_soft<<<(N + 255) / 256, 256, 0, stream>>>(H3, shot, outS, N, t, 16);
    }
  }
  (void)n_in; (void)out_size; (void)ws_size;
}
